// FocusedLinearAttention_26061861552487
// MI455X (gfx1250) — compile-verified, matrix operand order checked by reading, not run on hardware
//
#include <hip/hip_runtime.h>
#include <hip/hip_fp16.h>

typedef __attribute__((ext_vector_type(16))) _Float16 v16h;
typedef __attribute__((ext_vector_type(8)))  _Float16 v8h;
typedef __attribute__((ext_vector_type(8)))  float    v8f;
typedef __attribute__((ext_vector_type(4)))  float    v4f;
typedef __attribute__((ext_vector_type(4)))  int      v4i;

#define FLA_EPS 1e-5f
#define NB 64
#define LDK 264   // padded K-stride (halves) for transposed [col][K] LDS blocks

static __device__ __forceinline__ v8f wmma_f16(v16h a, v16h b, v8f c) {
  return __builtin_amdgcn_wmma_f32_16x16x32_f16(false, a, false, b, (short)0, c,
                                                false, false);
}

// Async 16-byte global -> LDS copy (GLOBAL_LOAD_ASYNC_TO_LDS_B128, ASYNCcnt).
static __device__ __forceinline__ void async_cp16(void* lds, const void* g) {
  __builtin_amdgcn_global_load_async_to_lds_b128(
      (__attribute__((address_space(1))) v4i*)g,
      (__attribute__((address_space(3))) v4i*)lds, 0, 0);
}

// WMMA 16x16x32 fragment from an f16 matrix whose fragment rows are contiguous
// in K (A row-major, or B when columns are contiguous in K — e.g. transposed
// LDS blocks and k^T from k stored [d][N]). Two 16B loads per lane.
// Lane L (0-15) holds row rowBase+L; lanes 16-31 same rows, upper K-halves:
// elems 0..7 = K k0+hi*8+{0..7}, elems 8..15 = K k0+16+hi*8+{0..7}.
static __device__ __forceinline__ v16h load_frag_f16(const _Float16* src,
                                                     int ld, int rowBase, int k0) {
  int lane = threadIdx.x & 31;
  const _Float16* p = src + (size_t)(rowBase + (lane & 15)) * ld + k0 + ((lane >> 4) << 3);
  v16h f;
#pragma unroll
  for (int j = 0; j < 8; ++j) {
    f[j]     = p[j];
    f[j + 8] = p[j + 16];
  }
  return f;
}

__global__ __launch_bounds__(256) void fla_zero_kernel(float* p, int n) {
  int i = blockIdx.x * 256 + threadIdx.x;
  if (i < n) p[i] = 0.f;
}

__global__ __launch_bounds__(256)
void fla_cvt_kernel(const float* __restrict__ s, _Float16* __restrict__ d, int n) {
  int i = blockIdx.x * 256 + threadIdx.x;
  if (i < n) d[i] = (_Float16)s[i];
}

// ---------------------------------------------------------------------------
// K1: qkv 1x1 conv GEMM + feature map (q,k) / passthrough (v) + ksum for kmean
// grid = (B*64, 3), block = 256 (8 waves). Each block: [256ch x 64pos], K=256.
// x block staged transposed into LDS as [n][c] f16 so B-frags are 2x16B reads.
// A = pre-converted f16 qkv weights (direct 16B fragment loads, L2-resident).
// ---------------------------------------------------------------------------
__global__ __launch_bounds__(256)
void fla_qkv_kernel(const float* __restrict__ x, const _Float16* __restrict__ qkv_wh,
                    const float* __restrict__ qkv_b,
                    _Float16* __restrict__ qf, _Float16* __restrict__ kf,
                    _Float16* __restrict__ vf, float* __restrict__ ksum)
{
  __shared__ _Float16 BshT[NB * LDK];
  __shared__ float colsum[NB];
  __shared__ float ksum_sh[256];

  const int tid   = threadIdx.x;
  const int b     = blockIdx.x >> 6;
  const int n0    = (blockIdx.x & 63) * NB;
  const int which = blockIdx.y;          // 0=q 1=k 2=v  (block-uniform)
  const int wave  = tid >> 5;
  const int lane  = tid & 31;
  const int ofs   = which << 8;          // row offset into qkv_w / qkv_b

  // Stage x[b, :, n0:n0+64] -> f16 LDS transposed (vector global loads)
  const float* xb = x + (size_t)b * 256 * 4096 + n0;
#pragma unroll
  for (int it = 0; it < 16; ++it) {
    int q = tid + it * 256;              // 0..4095 float4-chunks
    int c  = q >> 4;
    int nc = (q & 15) << 2;
    v4f xv = *(const v4f*)(xb + (size_t)c * 4096 + nc);
#pragma unroll
    for (int u = 0; u < 4; ++u) BshT[(nc + u) * LDK + c] = (_Float16)xv[u];
  }
  if (tid < NB) colsum[tid] = 0.f;
  ksum_sh[tid] = 0.f;
  __syncthreads();

  v8f acc[2][4] = {};
#pragma unroll 1
  for (int kk = 0; kk < 256; kk += 32) {
    v16h a0 = load_frag_f16(qkv_wh, 256, ofs + (wave * 2 + 0) * 16, kk);
    v16h a1 = load_frag_f16(qkv_wh, 256, ofs + (wave * 2 + 1) * 16, kk);
#pragma unroll
    for (int nt = 0; nt < 4; ++nt) {
      v16h bf = load_frag_f16(BshT, LDK, nt * 16, kk);
      acc[0][nt] = wmma_f16(a0, bf, acc[0][nt]);
      acc[1][nt] = wmma_f16(a1, bf, acc[1][nt]);
    }
  }

  const int r = lane & 15, hi = lane >> 4;
  if (which == 2) {                       // v: bias only, write f16
#pragma unroll
    for (int i = 0; i < 2; ++i)
#pragma unroll
      for (int j = 0; j < 8; ++j) {
        int m = (wave * 2 + i) * 16 + j + 8 * hi;
        float bias = qkv_b[ofs + m];
#pragma unroll
        for (int nt = 0; nt < 4; ++nt) {
          int n = n0 + nt * 16 + r;
          vf[(size_t)(b * 256 + m) * 4096 + n] = (_Float16)(acc[i][nt][j] + bias);
        }
      }
  } else {
    // feature map: t = (relu(val)+eps)^2, then column L2 normalize over 256 ch
#pragma unroll
    for (int i = 0; i < 2; ++i)
#pragma unroll
      for (int j = 0; j < 8; ++j) {
        int m = (wave * 2 + i) * 16 + j + 8 * hi;
        float bias = qkv_b[ofs + m];
#pragma unroll
        for (int nt = 0; nt < 4; ++nt) {
          float t = fmaxf(acc[i][nt][j] + bias, 0.f) + FLA_EPS;
          acc[i][nt][j] = t * t;
        }
      }
#pragma unroll
    for (int nt = 0; nt < 4; ++nt) {
      float s = 0.f;
#pragma unroll
      for (int i = 0; i < 2; ++i)
#pragma unroll
        for (int j = 0; j < 8; ++j) s += acc[i][nt][j] * acc[i][nt][j];
      atomicAdd(&colsum[nt * 16 + r], s);     // ds_add_f32
    }
    __syncthreads();
    float inv[4];
#pragma unroll
    for (int nt = 0; nt < 4; ++nt)
      inv[nt] = 1.f / fmaxf(sqrtf(colsum[nt * 16 + r]), 1e-12f);
    _Float16* dst = (which == 0) ? qf : kf;
#pragma unroll
    for (int i = 0; i < 2; ++i)
#pragma unroll
      for (int j = 0; j < 8; ++j) {
        int m = (wave * 2 + i) * 16 + j + 8 * hi;
        float rowsum = 0.f;
#pragma unroll
        for (int nt = 0; nt < 4; ++nt) {
          float tn = acc[i][nt][j] * inv[nt];
          dst[(size_t)(b * 256 + m) * 4096 + n0 + nt * 16 + r] = (_Float16)tn;
          rowsum += tn;
        }
        if (which == 1) atomicAdd(&ksum_sh[m], rowsum);
      }
    if (which == 1) {
      __syncthreads();
      atomicAdd(&ksum[b * 256 + tid], ksum_sh[tid]);
    }
  }
}

// ---------------------------------------------------------------------------
// K2: vkT = v @ k^T  [256x256], K=N=4096, split-K across 16 blocks per batch.
// grid = B*16, block = 512 (16 waves). k^T chunks double-buffered via async
// LDS copies (ASYNCcnt in-order completion: wait<=2 keeps newest in flight).
// ---------------------------------------------------------------------------
__global__ __launch_bounds__(512)
void fla_vkt_kernel(const _Float16* __restrict__ vf, const _Float16* __restrict__ kf,
                    float* __restrict__ vkT)
{
  __shared__ _Float16 Bsh[2][256 * 40];  // [buf][col j][k-chunk 32, pad 40]
  const int tid  = threadIdx.x;
  const int wave = tid >> 5;
  const int lane = tid & 31;
  const int b    = blockIdx.x >> 4;
  const int k0g  = (blockIdx.x & 15) * 256;

  const _Float16* vb = vf + (size_t)b * 256 * 4096;
  const _Float16* kb = kf + (size_t)b * 256 * 4096 + k0g;

  const int j0 = tid >> 2, piece0 = (tid & 3) << 3;       // chunk 0 of 2
  const int j1 = (tid + 512) >> 2, piece1 = (tid & 3) << 3;

  // prologue: issue k-chunk 0 into buf 0
  async_cp16(&Bsh[0][j0 * 40 + piece0], kb + (size_t)j0 * 4096 + piece0);
  async_cp16(&Bsh[0][j1 * 40 + piece1], kb + (size_t)j1 * 4096 + piece1);

  v8f acc[16] = {};
#pragma unroll 1
  for (int step = 0; step < 8; ++step) {
    int kk = step * 32;
    if (step < 7) {                      // issue next chunk into other buffer
      async_cp16(&Bsh[(step + 1) & 1][j0 * 40 + piece0],
                 kb + (size_t)j0 * 4096 + kk + 32 + piece0);
      async_cp16(&Bsh[(step + 1) & 1][j1 * 40 + piece1],
                 kb + (size_t)j1 * 4096 + kk + 32 + piece1);
      __builtin_amdgcn_s_wait_asynccnt(2);   // chunk `step` landed (in-order)
    } else {
      __builtin_amdgcn_s_wait_asynccnt(0);
    }
    __syncthreads();
    v16h a = load_frag_f16(vb, 4096, wave * 16, k0g + kk);
    const _Float16* bsrc = Bsh[step & 1];
#pragma unroll
    for (int nt = 0; nt < 16; ++nt) {
      v16h bf = load_frag_f16(bsrc, 40, nt * 16, 0);  // cols contiguous in K
      acc[nt] = wmma_f16(a, bf, acc[nt]);
    }
    __syncthreads();                     // all waves done before buf reuse
  }
  const int r = lane & 15, hi = lane >> 4;
#pragma unroll
  for (int nt = 0; nt < 16; ++nt)
#pragma unroll
    for (int j = 0; j < 8; ++j) {
      int m = wave * 16 + j + 8 * hi;
      atomicAdd(&vkT[((size_t)b * 256 + m) * 256 + nt * 16 + r], acc[nt][j]);
    }
}

// ---------------------------------------------------------------------------
// K4: depthwise 5x5 SAME conv on v, f16 tile staged via async LDS copies.
// grid = B*256*16, block = 256 (4 rows of 64 pixels + 2-row halo per block)
// ---------------------------------------------------------------------------
__global__ __launch_bounds__(256)
void fla_dwconv_kernel(const _Float16* __restrict__ vf, const float* __restrict__ w5,
                       const float* __restrict__ wb, _Float16* __restrict__ dwv)
{
  __shared__ _Float16 tile[8 * 64];      // rows h0-2 .. h0+5
  int t  = blockIdx.x & 15;
  int bc = blockIdx.x >> 4;              // b*256 + c
  int c  = bc & 255;
  int h0 = t * 4;
  const _Float16* src = vf + (size_t)bc * 4096;
  if (threadIdx.x < 64) {
    int row = threadIdx.x >> 3;
    int ch  = (threadIdx.x & 7) << 3;
    int hh  = h0 + row - 2;
    if (hh >= 0 && hh < 64)
      async_cp16(&tile[row * 64 + ch], src + hh * 64 + ch);
  }
  __builtin_amdgcn_s_wait_asynccnt(0);
  __syncthreads();

  int p = t * 256 + threadIdx.x;
  int h = p >> 6, wc = p & 63;
  const float* wt = w5 + c * 25;
  float acc = wb[c];
#pragma unroll
  for (int dy = -2; dy <= 2; ++dy) {
    int hh = h + dy;
    if (hh < 0 || hh > 63) continue;
    int lr = hh - h0 + 2;
#pragma unroll
    for (int dx = -2; dx <= 2; ++dx) {
      int ww = wc + dx;
      if (ww < 0 || ww > 63) continue;
      acc += wt[(dy + 2) * 5 + (dx + 2)] * (float)tile[lr * 64 + ww];
    }
  }
  dwv[(size_t)bc * 4096 + p] = (_Float16)acc;
}

// ---------------------------------------------------------------------------
// K5: attn = (vkT @ q) * (1/4096) / z ; RMSNorm * norm_w ; + dw -> y (f16)
// grid = B*64, block = 256. q block staged transposed [n][c].
// ---------------------------------------------------------------------------
__global__ __launch_bounds__(256)
void fla_attn_kernel(const _Float16* __restrict__ vkTh, const _Float16* __restrict__ qf,
                     const float* __restrict__ ksum, const _Float16* __restrict__ dwv,
                     const float* __restrict__ norm_w, _Float16* __restrict__ y)
{
  __shared__ _Float16 BshT[NB * LDK];
  __shared__ float kmean_sh[256];
  __shared__ float zcol[NB];
  __shared__ float colss[NB];

  const int tid  = threadIdx.x;
  const int b    = blockIdx.x >> 6;
  const int n0   = (blockIdx.x & 63) * NB;
  const int wave = tid >> 5;
  const int lane = tid & 31;

  const _Float16* qb = qf + (size_t)b * 256 * 4096 + n0;
#pragma unroll
  for (int it = 0; it < 8; ++it) {
    int q = tid + it * 256;              // 0..2047 16B-chunks
    int cch = q >> 3;
    int nc  = (q & 7) << 3;
    v8h hv = *(const v8h*)(qb + (size_t)cch * 4096 + nc);
#pragma unroll
    for (int u = 0; u < 8; ++u) BshT[(nc + u) * LDK + cch] = hv[u];
  }
  kmean_sh[tid] = ksum[b * 256 + tid] * (1.f / 4096.f);
  if (tid < NB) { colss[tid] = 0.f; zcol[tid] = FLA_EPS; }
  __syncthreads();

  {                                       // z = kmean . q(col) + eps, 4-way split
    int col = tid & 63, quarter = tid >> 6;
    const _Float16* bp = BshT + col * LDK + quarter * 64;
    const float* kp = kmean_sh + quarter * 64;
    float s = 0.f;
#pragma unroll
    for (int cc = 0; cc < 64; cc += 8) {
      v8h hv = *(const v8h*)(bp + cc);
#pragma unroll
      for (int u = 0; u < 8; ++u) s += kp[cc + u] * (float)hv[u];
    }
    atomicAdd(&zcol[col], s);
  }

  const _Float16* A = vkTh + (size_t)b * 256 * 256;
  v8f acc[2][4] = {};
#pragma unroll 1
  for (int kk = 0; kk < 256; kk += 32) {
    v16h a0 = load_frag_f16(A, 256, (wave * 2 + 0) * 16, kk);
    v16h a1 = load_frag_f16(A, 256, (wave * 2 + 1) * 16, kk);
#pragma unroll
    for (int nt = 0; nt < 4; ++nt) {
      v16h bf = load_frag_f16(BshT, LDK, nt * 16, kk);
      acc[0][nt] = wmma_f16(a0, bf, acc[0][nt]);
      acc[1][nt] = wmma_f16(a1, bf, acc[1][nt]);
    }
  }
  __syncthreads();                        // zcol complete, colss zeroed
  const int r = lane & 15, hi = lane >> 4;
#pragma unroll
  for (int nt = 0; nt < 4; ++nt) {
    float z = zcol[nt * 16 + r];
    float s = 0.f;
#pragma unroll
    for (int i = 0; i < 2; ++i)
#pragma unroll
      for (int j = 0; j < 8; ++j) {
        float a = acc[i][nt][j] * (1.f / 4096.f) / z;
        acc[i][nt][j] = a;
        s += a * a;
      }
    atomicAdd(&colss[nt * 16 + r], s);
  }
  __syncthreads();
  float rinv[4];
#pragma unroll
  for (int nt = 0; nt < 4; ++nt)
    rinv[nt] = 1.f / sqrtf(colss[nt * 16 + r] * (1.f / 256.f) + FLA_EPS);
#pragma unroll
  for (int i = 0; i < 2; ++i)
#pragma unroll
    for (int j = 0; j < 8; ++j) {
      int m = (wave * 2 + i) * 16 + j + 8 * hi;
      float nw = norm_w[m];
#pragma unroll
      for (int nt = 0; nt < 4; ++nt) {
        size_t idx = (size_t)(b * 256 + m) * 4096 + n0 + nt * 16 + r;
        float yv = acc[i][nt][j] * rinv[nt] * nw + (float)dwv[idx];
        y[idx] = (_Float16)yv;
      }
    }
}

// ---------------------------------------------------------------------------
// K6: out = proj_w @ y + proj_b (fp32 out). grid = B*64, block = 256
// A = pre-converted f16 proj weights.
// ---------------------------------------------------------------------------
__global__ __launch_bounds__(256)
void fla_proj_kernel(const _Float16* __restrict__ y, const _Float16* __restrict__ proj_wh,
                     const float* __restrict__ proj_b, float* __restrict__ out)
{
  __shared__ _Float16 BshT[NB * LDK];
  const int tid  = threadIdx.x;
  const int b    = blockIdx.x >> 6;
  const int n0   = (blockIdx.x & 63) * NB;
  const int wave = tid >> 5;
  const int lane = tid & 31;

  const _Float16* yb = y + (size_t)b * 256 * 4096 + n0;
#pragma unroll
  for (int it = 0; it < 8; ++it) {
    int q = tid + it * 256;
    int cch = q >> 3;
    int nc  = (q & 7) << 3;
    v8h hv = *(const v8h*)(yb + (size_t)cch * 4096 + nc);
#pragma unroll
    for (int u = 0; u < 8; ++u) BshT[(nc + u) * LDK + cch] = hv[u];
  }
  __syncthreads();

  v8f acc[2][4] = {};
#pragma unroll 1
  for (int kk = 0; kk < 256; kk += 32) {
    v16h a0 = load_frag_f16(proj_wh, 256, (wave * 2 + 0) * 16, kk);
    v16h a1 = load_frag_f16(proj_wh, 256, (wave * 2 + 1) * 16, kk);
#pragma unroll
    for (int nt = 0; nt < 4; ++nt) {
      v16h bf = load_frag_f16(BshT, LDK, nt * 16, kk);
      acc[0][nt] = wmma_f16(a0, bf, acc[0][nt]);
      acc[1][nt] = wmma_f16(a1, bf, acc[1][nt]);
    }
  }
  const int r = lane & 15, hi = lane >> 4;
#pragma unroll
  for (int i = 0; i < 2; ++i)
#pragma unroll
    for (int j = 0; j < 8; ++j) {
      int m = (wave * 2 + i) * 16 + j + 8 * hi;
      float bias = proj_b[m];
#pragma unroll
      for (int nt = 0; nt < 4; ++nt)
        out[(size_t)(b * 256 + m) * 4096 + n0 + nt * 16 + r] = acc[i][nt][j] + bias;
    }
}

extern "C" void kernel_launch(void* const* d_in, const int* in_sizes, int n_in,
                              void* d_out, int out_size, void* d_ws, size_t ws_size,
                              hipStream_t stream)
{
  (void)in_sizes; (void)n_in; (void)out_size; (void)ws_size;
  const float* x      = (const float*)d_in[0];
  const float* qkv_w  = (const float*)d_in[1];
  const float* qkv_b  = (const float*)d_in[2];
  const float* dwc_w  = (const float*)d_in[3];
  const float* dwc_b  = (const float*)d_in[4];
  const float* norm_w = (const float*)d_in[5];
  const float* proj_w = (const float*)d_in[6];
  const float* proj_b = (const float*)d_in[7];
  float* out = (float*)d_out;

  const size_t B = 16, D = 256, N = 4096;
  char* ws = (char*)d_ws;
  size_t off = 0;
  auto take = [&](size_t bytes) {
    char* p = ws + off;
    off = (off + bytes + 255) & ~(size_t)255;
    return p;
  };
  _Float16* qf      = (_Float16*)take(B * D * N * 2);
  _Float16* kf      = (_Float16*)take(B * D * N * 2);   // reused as y later
  _Float16* vf      = (_Float16*)take(B * D * N * 2);
  _Float16* dwv     = (_Float16*)take(B * D * N * 2);
  float*    vkT     = (float*)   take(B * D * D * 4);
  _Float16* vkTh    = (_Float16*)take(B * D * D * 2);
  float*    ksum    = (float*)   take(B * D * 4);
  _Float16* qkv_wh  = (_Float16*)take(3 * D * D * 2);
  _Float16* proj_wh = (_Float16*)take(D * D * 2);

  int nvkt = (int)(B * D * D);
  int nqkvw = (int)(3 * D * D);
  int nprojw = (int)(D * D);
  fla_zero_kernel<<<dim3((nvkt + 255) / 256), 256, 0, stream>>>(vkT, nvkt);
  fla_zero_kernel<<<dim3((int)((B * D + 255) / 256)), 256, 0, stream>>>(ksum, (int)(B * D));
  fla_cvt_kernel<<<dim3((nqkvw + 255) / 256), 256, 0, stream>>>(qkv_w, qkv_wh, nqkvw);
  fla_cvt_kernel<<<dim3((nprojw + 255) / 256), 256, 0, stream>>>(proj_w, proj_wh, nprojw);
  fla_qkv_kernel<<<dim3(1024, 3), 256, 0, stream>>>(x, qkv_wh, qkv_b, qf, kf, vf, ksum);
  fla_vkt_kernel<<<dim3(256), 512, 0, stream>>>(vf, kf, vkT);
  fla_cvt_kernel<<<dim3((nvkt + 255) / 256), 256, 0, stream>>>(vkT, vkTh, nvkt);
  fla_dwconv_kernel<<<dim3(65536), 256, 0, stream>>>(vf, dwc_w, dwc_b, dwv);
  _Float16* yb = kf;  // kf dead after vkt; reuse for y
  fla_attn_kernel<<<dim3(1024), 256, 0, stream>>>(vkTh, qf, ksum, dwv, norm_w, yb);
  fla_proj_kernel<<<dim3(1024), 256, 0, stream>>>(yb, proj_w ? proj_wh : proj_wh, proj_b, out);
}
